// MoMoShareLayer_60524679135402
// MI455X (gfx1250) — compile-verified, matrix operand order checked by reading, not run on hardware
//
#include <hip/hip_runtime.h>
#include <hip/hip_bf16.h>

// ---------------------------------------------------------------------------
// MoMo shared/unique expert layer for gfx1250 (MI455X), wave32 + WMMA f16.
// fp32 in memory, f16 operands, f32 WMMA accumulation.
// ---------------------------------------------------------------------------

typedef __attribute__((ext_vector_type(16))) _Float16 v16h;
typedef __attribute__((ext_vector_type(8)))  _Float16 v8h;
typedef __attribute__((ext_vector_type(4)))  _Float16 v4h;
typedef __attribute__((ext_vector_type(8)))  float    v8f;

#define Bdim 64
#define Sdim 256
#define Hdim 768
#define Rdim 128
#define NHn  12
#define DHd  64
#define BS   (Bdim * Sdim)   // 16384 tokens

#define F_GELU  1
#define F_ACCUM 2

// ---------------- WMMA fragment loaders (ISA 7.12.2 layouts) ----------------
// A/B frag, 16 (rows) x 32 (K), lane l<16: row l, K {0..7,16..23};
// lane l>=16: row l-16, K {8..15,24..31}. Row-major source with stride ld.

__device__ inline v16h frag_f16s(const _Float16* base, int ld) {
  int lane = threadIdx.x & 31;
  int l = lane & 15;
  int kh = (lane >> 4) << 3;           // 0 or 8
  const _Float16* p = base + l * ld + kh;
  v8h g0 = *(const v8h*)(p);           // K kh..kh+7   -> ds_load_b128
  v8h g1 = *(const v8h*)(p + 16);      // K kh+16..kh+23
  v16h f;
#pragma unroll
  for (int i = 0; i < 8; ++i) { f[i] = g0[i]; f[8 + i] = g1[i]; }
  return f;
}

__device__ inline v16h frag_f32g(const float* __restrict__ base, int ld) {
  int lane = threadIdx.x & 31;
  int l = lane & 15;
  int kh = (lane >> 4) << 3;
  const float* p = base + (size_t)l * ld + kh;
  float4 x0 = *(const float4*)(p + 0);
  float4 x1 = *(const float4*)(p + 4);
  float4 x2 = *(const float4*)(p + 16);
  float4 x3 = *(const float4*)(p + 20);
  v16h f;
  f[0]  = (_Float16)x0.x; f[1]  = (_Float16)x0.y; f[2]  = (_Float16)x0.z; f[3]  = (_Float16)x0.w;
  f[4]  = (_Float16)x1.x; f[5]  = (_Float16)x1.y; f[6]  = (_Float16)x1.z; f[7]  = (_Float16)x1.w;
  f[8]  = (_Float16)x2.x; f[9]  = (_Float16)x2.y; f[10] = (_Float16)x2.z; f[11] = (_Float16)x2.w;
  f[12] = (_Float16)x3.x; f[13] = (_Float16)x3.y; f[14] = (_Float16)x3.z; f[15] = (_Float16)x3.w;
  return f;
}

// branch-free tanh-GELU: tanh(u) = sign(u) * (1 - 2/(exp(2|u|)+1))
__device__ inline float gelu_tanh(float v) {
  float u  = 0.7978845608028654f * (v + 0.044715f * v * v * v);
  float t  = __expf(-2.0f * fabsf(u));          // (0,1], never overflows
  float th = (1.0f - t) / (1.0f + t);
  th = copysignf(th, u);
  return 0.5f * v * (1.0f + th);
}

// ---------------- generic WMMA GEMM: C[M,N] = act(A[M,K] @ B[K,N] + bias) ---
// block = 256 threads (8 waves), tile 128(M) x 128(N), K step 32,
// double-buffered LDS (one barrier per K step).
// Waves arranged 4(M) x 2(N); each wave computes 32x64 via 2x4 16x16x32 frags
// -> 8 WMMA per 6 fragment loads per K step.
template <int FLAGS, bool HAS_BIAS>
__global__ __launch_bounds__(256) void gemm_wmma_kernel(
    const float* __restrict__ A, int lda,
    const float* __restrict__ Bw, int ldb,
    float* __restrict__ C, int ldc,
    const float* __restrict__ bias, int K)
{
  __shared__ __align__(16) _Float16 As[2][128 * 32];   // 2 x 8 KB
  __shared__ __align__(16) _Float16 Bs[2][128 * 32];   // 2 x 8 KB (transposed [n][k])
  const int tid  = threadIdx.x;
  const int wid  = tid >> 5;
  const int lane = tid & 31;
  const int m0 = blockIdx.y * 128;
  const int n0 = blockIdx.x * 128;
  const int wm = (wid >> 1) * 32;
  const int wn = (wid & 1) * 64;

  // A staging: row ar, 16 contiguous cols at ac (2 x v8h LDS stores)
  const int ar = tid >> 1;             // 0..127
  const int ac = (tid & 1) * 16;       // 0 or 16
  // B staging: 4x4 register micro-transpose -> 4 x v4h LDS stores
  const int kb = (tid & 7) * 4;        // 0..28
  const int nb = (tid >> 3) * 4;       // 0..124

  float4 a_reg[4];
  float4 b_reg[4];

  auto load_tile = [&](int kk) {
    const float* pa = A + (size_t)(m0 + ar) * lda + kk + ac;
    a_reg[0] = *(const float4*)(pa + 0);
    a_reg[1] = *(const float4*)(pa + 4);
    a_reg[2] = *(const float4*)(pa + 8);
    a_reg[3] = *(const float4*)(pa + 12);
#pragma unroll
    for (int dk = 0; dk < 4; ++dk)
      b_reg[dk] = *(const float4*)(Bw + (size_t)(kk + kb + dk) * ldb + n0 + nb);
  };
  auto store_tile = [&](int buf) {
    const float* af = (const float*)a_reg;
    v8h h0, h1;
#pragma unroll
    for (int i = 0; i < 8; ++i) { h0[i] = (_Float16)af[i]; h1[i] = (_Float16)af[8 + i]; }
    *(v8h*)(&As[buf][ar * 32 + ac + 0]) = h0;
    *(v8h*)(&As[buf][ar * 32 + ac + 8]) = h1;
    const float* bf = (const float*)b_reg;   // bf[dk*4 + dn]
#pragma unroll
    for (int dn = 0; dn < 4; ++dn) {
      v4h q;
#pragma unroll
      for (int dk = 0; dk < 4; ++dk) q[dk] = (_Float16)bf[dk * 4 + dn];
      *(v4h*)(&Bs[buf][(nb + dn) * 32 + kb]) = q;
    }
  };

  v8f acc[2][4] = {};
  load_tile(0);
  int buf = 0;
  for (int kk = 0; kk < K; kk += 32) {
    store_tile(buf);
    __syncthreads();
    if (kk + 32 < K) {
      if (kk + 64 < K) {   // prefetch two tiles ahead -> global_prefetch_b8
        __builtin_prefetch(A + (size_t)(m0 + ar) * lda + kk + 64 + ac, 0, 1);
        __builtin_prefetch(Bw + (size_t)(kk + 64 + kb) * ldb + n0 + nb, 0, 1);
      }
      load_tile(kk + 32);
    }
    v16h a0 = frag_f16s(&As[buf][(wm + 0) * 32], 32);
    v16h a1 = frag_f16s(&As[buf][(wm + 16) * 32], 32);
#pragma unroll
    for (int ni = 0; ni < 4; ++ni) {
      v16h b = frag_f16s(&Bs[buf][(wn + ni * 16) * 32], 32);
      acc[0][ni] = __builtin_amdgcn_wmma_f32_16x16x32_f16(false, a0, false, b, (short)0, acc[0][ni], false, false);
      acc[1][ni] = __builtin_amdgcn_wmma_f32_16x16x32_f16(false, a1, false, b, (short)0, acc[1][ni], false, false);
    }
    buf ^= 1;
  }

  // C/D layout: vgpr r, lanes 0-15 -> row r, lanes 16-31 -> row r+8; col = lane&15
  const int cn = lane & 15;
  const int ro = (lane >> 4) << 3;
#pragma unroll
  for (int ni = 0; ni < 4; ++ni) {
    int n = n0 + wn + ni * 16 + cn;
    float bv = 0.0f;
    if constexpr (HAS_BIAS) bv = bias[n];
#pragma unroll
    for (int mi = 0; mi < 2; ++mi) {
#pragma unroll
      for (int r = 0; r < 8; ++r) {
        int m = m0 + wm + mi * 16 + ro + r;
        float v = acc[mi][ni][r] + bv;
        if constexpr ((FLAGS & F_GELU) != 0) v = gelu_tanh(v);
        size_t off = (size_t)m * ldc + n;
        if constexpr ((FLAGS & F_ACCUM) != 0) v += C[off];
        C[off] = v;
      }
    }
  }
}

// ---------------- fused attention per (b, head, 32-query tile) --------------
// phase1: S = (Q/8) K^T + mask (WMMA, K frags straight from global f32)
// phase2: row softmax in LDS -> f16 P (lane owns 8 contiguous cols)
// phase3: O = P V (WMMA, V transposed into LDS aliasing dead score buffer)
__global__ __launch_bounds__(256) void attn_kernel(
    const float* __restrict__ Q, const float* __restrict__ Kx,
    const float* __restrict__ V, const float* __restrict__ mask,
    float* __restrict__ CTX)
{
  __shared__ __align__(16) char smem[53248];          // 52 KB static
  _Float16* Qs = (_Float16*)smem;                     // [32][64]   4 KB
  float*    Sb = (float*)(smem + 4096);               // [32][256] 32 KB
  _Float16* Pb = (_Float16*)(smem + 36864);           // [32][256] 16 KB
  _Float16* Vt = (_Float16*)(smem + 4096);            // [64][256] 32 KB (aliases Sb)

  const int qt = blockIdx.x, hI = blockIdx.y, bI = blockIdx.z;
  const int q0 = qt * 32;
  const size_t rowbase = ((size_t)bI * Sdim + q0) * Hdim + hI * DHd;
  const float* Qg = Q + rowbase;
  const float* Kg = Kx + (size_t)bI * Sdim * Hdim + hI * DHd;
  const float* Vg = V  + (size_t)bI * Sdim * Hdim + hI * DHd;
  const int tid = threadIdx.x;
  const int wid = tid >> 5, lane = tid & 31;
  const int cn = lane & 15, ro = (lane >> 4) << 3;

  { // stage Q (fold 1/sqrt(64)) as one v8h store per thread
    int r = tid >> 3, c = (tid & 7) * 8;
    const float* p = Qg + (size_t)r * Hdim + c;
    float4 x0 = *(const float4*)(p);
    float4 x1 = *(const float4*)(p + 4);
    v8h h;
    h[0] = (_Float16)(x0.x * 0.125f); h[1] = (_Float16)(x0.y * 0.125f);
    h[2] = (_Float16)(x0.z * 0.125f); h[3] = (_Float16)(x0.w * 0.125f);
    h[4] = (_Float16)(x1.x * 0.125f); h[5] = (_Float16)(x1.y * 0.125f);
    h[6] = (_Float16)(x1.z * 0.125f); h[7] = (_Float16)(x1.w * 0.125f);
    *(v8h*)(&Qs[r * 64 + c]) = h;
  }
  __syncthreads();

  { // phase 1: waves 2(q) x 4(k); each 16q x 64k
    const int wq = wid >> 2;
    const int wk = wid & 3;
    v8f acc[4] = {};
#pragma unroll
    for (int kk = 0; kk < 64; kk += 32) {
      v16h a = frag_f16s(Qs + (wq * 16) * 64 + kk, 64);
#pragma unroll
      for (int ni = 0; ni < 4; ++ni) {
        v16h bf = frag_f32g(Kg + (size_t)(wk * 64 + ni * 16) * Hdim + kk, Hdim);
        acc[ni] = __builtin_amdgcn_wmma_f32_16x16x32_f16(false, a, false, bf, (short)0, acc[ni], false, false);
      }
    }
#pragma unroll
    for (int ni = 0; ni < 4; ++ni) {
      int key = wk * 64 + ni * 16 + cn;
      float mv = mask[(size_t)bI * Sdim + key];
#pragma unroll
      for (int r = 0; r < 8; ++r)
        Sb[(wq * 16 + ro + r) * 256 + key] = acc[ni][r] + mv;
    }
  }
  __syncthreads();

  { // phase 2: softmax; wave owns 4 rows; lane owns 8 contiguous cols
#pragma unroll
    for (int rr = 0; rr < 4; ++rr) {
      int row = wid * 4 + rr;
      const float* sr = Sb + row * 256 + lane * 8;
      float ev[8];
      float mx = -3.0e38f;
#pragma unroll
      for (int j = 0; j < 8; ++j) { ev[j] = sr[j]; mx = fmaxf(mx, ev[j]); }
#pragma unroll
      for (int o = 16; o > 0; o >>= 1) mx = fmaxf(mx, __shfl_xor(mx, o, 32));
      float sum = 0.0f;
#pragma unroll
      for (int j = 0; j < 8; ++j) { ev[j] = __expf(ev[j] - mx); sum += ev[j]; }
#pragma unroll
      for (int o = 16; o > 0; o >>= 1) sum += __shfl_xor(sum, o, 32);
      float inv = 1.0f / sum;
      v8h h;
#pragma unroll
      for (int j = 0; j < 8; ++j) h[j] = (_Float16)(ev[j] * inv);
      *(v8h*)(&Pb[row * 256 + lane * 8]) = h;
    }
  }
  __syncthreads();

  { // stage V transposed via 4-wide micro-transpose, v8h stores
    int dg = (tid & 15) * 4;
    int sg = (tid >> 4) * 16;
#pragma unroll
    for (int sh = 0; sh < 16; sh += 8) {
      float4 col[8];
#pragma unroll
      for (int ds = 0; ds < 8; ++ds)
        col[ds] = *(const float4*)(Vg + (size_t)(sg + sh + ds) * Hdim + dg);
      const float* cf = (const float*)col;
#pragma unroll
      for (int dn = 0; dn < 4; ++dn) {
        v8h h;
#pragma unroll
        for (int ds = 0; ds < 8; ++ds) h[ds] = (_Float16)cf[ds * 4 + dn];
        *(v8h*)(&Vt[(dg + dn) * 256 + sg + sh]) = h;
      }
    }
  }
  __syncthreads();

  { // phase 3: waves 2(q) x 4(d); each 16q x 16d, K loop over 256 keys.
    // Two interleaved accumulators break the WMMA->WMMA RAW hazard chain
    // (5-NOP penalty for f16 per ISA 7.12.1).
    const int wq = wid >> 2;
    const int wd = wid & 3;
    v8f acc0 = {}, acc1 = {};
#pragma unroll
    for (int kk = 0; kk < 256; kk += 64) {
      v16h a0 = frag_f16s(Pb + (wq * 16) * 256 + kk, 256);
      v16h b0 = frag_f16s(Vt + (wd * 16) * 256 + kk, 256);
      acc0 = __builtin_amdgcn_wmma_f32_16x16x32_f16(false, a0, false, b0, (short)0, acc0, false, false);
      v16h a1 = frag_f16s(Pb + (wq * 16) * 256 + kk + 32, 256);
      v16h b1 = frag_f16s(Vt + (wd * 16) * 256 + kk + 32, 256);
      acc1 = __builtin_amdgcn_wmma_f32_16x16x32_f16(false, a1, false, b1, (short)0, acc1, false, false);
    }
    float* Cg = CTX + rowbase;
    int d = wd * 16 + cn;
#pragma unroll
    for (int r = 0; r < 8; ++r)
      Cg[(size_t)(wq * 16 + ro + r) * Hdim + d] = acc0[r] + acc1[r];
  }
}

// ---------------- block mean/var reduce (H=768, 256 threads) ----------------
__device__ inline void block_meanvar(float s, float s2, int tid, float* mean_o, float* rinv_o) {
  __shared__ float rs[8], rs2[8];
#pragma unroll
  for (int o = 16; o > 0; o >>= 1) { s += __shfl_xor(s, o, 32); s2 += __shfl_xor(s2, o, 32); }
  int wid = tid >> 5, lane = tid & 31;
  if (lane == 0) { rs[wid] = s; rs2[wid] = s2; }
  __syncthreads();
  float a  = rs[lane & 7];
  float a2 = rs2[lane & 7];
#pragma unroll
  for (int o = 4; o > 0; o >>= 1) { a += __shfl_xor(a, o, 32); a2 += __shfl_xor(a2, o, 32); }
  float mean = a * (1.0f / 768.0f);
  float var  = a2 * (1.0f / 768.0f) - mean * mean;
  *mean_o = mean;
  *rinv_o = rsqrtf(var + 1e-12f);
}

// Out = LN(X + Y)
__global__ __launch_bounds__(256) void add_ln_kernel(
    const float* __restrict__ Xr, const float* __restrict__ Y,
    const float* __restrict__ g, const float* __restrict__ bb,
    float* __restrict__ Out)
{
  size_t t = blockIdx.x;
  int tid = threadIdx.x;
  const float* x = Xr + t * Hdim;
  const float* y = Y + t * Hdim;
  float v[3], s = 0.0f, s2 = 0.0f;
#pragma unroll
  for (int i = 0; i < 3; ++i) { int j = tid + i * 256; float z = x[j] + y[j]; v[i] = z; s += z; s2 += z * z; }
  float mean, rinv;
  block_meanvar(s, s2, tid, &mean, &rinv);
#pragma unroll
  for (int i = 0; i < 3; ++i) { int j = tid + i * 256; Out[t * Hdim + j] = (v[i] - mean) * rinv * g[j] + bb[j]; }
}

// token-level switch-FFN router: 4 logits -> softmax -> (idx, pmax).
// Router weights (12 KB f32) staged global->LDS with CDNA5 async-to-LDS copies
// (ASYNCcnt-tracked, no VGPR round trip), then s_wait_asynccnt + barrier.
__global__ __launch_bounds__(256) void ffn_router_kernel(
    const float* __restrict__ Xa, const float* __restrict__ rw, const float* __restrict__ rb,
    int* __restrict__ idx, float* __restrict__ pmax)
{
  __shared__ __align__(16) float wsm[Hdim * 4];
  int tid = threadIdx.x;
  {
    int fi = tid * 12;                      // 12 floats = 3 x b128 per thread
#pragma unroll
    for (int c = 0; c < 3; ++c) {
      unsigned ldsa = (unsigned)(size_t)(&wsm[fi + c * 4]);
      unsigned long long ga = (unsigned long long)(size_t)(rw + fi + c * 4);
      asm volatile("global_load_async_to_lds_b128 %0, %1, off"
                   :: "v"(ldsa), "v"(ga) : "memory");
    }
    asm volatile("s_wait_asynccnt 0x0" ::: "memory");
  }
  __syncthreads();
  size_t t = (size_t)blockIdx.x * 256 + tid;
  const float* x = Xa + t * Hdim;
  float l0 = rb[0], l1 = rb[1], l2 = rb[2], l3 = rb[3];
  for (int j = 0; j < Hdim; ++j) {
    float xv = x[j];
    l0 += xv * wsm[j * 4 + 0]; l1 += xv * wsm[j * 4 + 1];
    l2 += xv * wsm[j * 4 + 2]; l3 += xv * wsm[j * 4 + 3];
  }
  float mm = fmaxf(fmaxf(l0, l1), fmaxf(l2, l3));
  float e0 = __expf(l0 - mm), e1 = __expf(l1 - mm), e2 = __expf(l2 - mm), e3 = __expf(l3 - mm);
  float sum = e0 + e1 + e2 + e3;
  int ii = 0; float bl = l0;
  if (l1 > bl) { bl = l1; ii = 1; }
  if (l2 > bl) { bl = l2; ii = 2; }
  if (l3 > bl) { bl = l3; ii = 3; }
  idx[t] = ii;
  pmax[t] = __expf(bl - mm) / sum;
}

// zero non-selected expert hidden slabs, scale selected by pmax (folds top-1)
__global__ __launch_bounds__(256) void hcat_mask_kernel(
    float* __restrict__ Hc, const int* __restrict__ fidx, const float* __restrict__ fpmax)
{
  size_t i = (size_t)blockIdx.x * 256 + threadIdx.x;   // over BS*512
  size_t t = i >> 9;
  int e = (int)((i >> 7) & 3);
  float sc = (e == fidx[t]) ? fpmax[t] : 0.0f;
  Hc[i] *= sc;
}

// y = LN(att + ffn + pmax*b2[sel]); mode0: Out=y (common); mode1: Out += y on routed rows
__global__ __launch_bounds__(256) void final_ln_combine_kernel(
    const float* __restrict__ ATT, const float* __restrict__ FFN,
    const float* __restrict__ b2, const int* __restrict__ fidx, const float* __restrict__ fpmax,
    const float* __restrict__ g, const float* __restrict__ bb,
    const int* __restrict__ routes, const float* __restrict__ pmaxB,
    int mode, int eidx, float* __restrict__ Out)
{
  size_t t = blockIdx.x;
  int tid = threadIdx.x;
  int e = fidx[t];
  float pm = fpmax[t];
  const float* a = ATT + t * Hdim;
  const float* f = FFN + t * Hdim;
  const float* be = b2 + (size_t)e * Hdim;
  float v[3], s = 0.0f, s2 = 0.0f;
#pragma unroll
  for (int i = 0; i < 3; ++i) { int j = tid + i * 256; float z = a[j] + f[j] + pm * be[j]; v[i] = z; s += z; s2 += z * z; }
  float mean, rinv;
  block_meanvar(s, s2, tid, &mean, &rinv);
  int bi = (int)(t >> 8);
  bool sel = (mode == 0) || (routes[bi] == eidx);
  float sc = 1.0f;
  if (mode != 0) { float p = pmaxB[bi]; sc = p / p; }  // reference scale = pmax/stop_grad(pmax)
#pragma unroll
  for (int i = 0; i < 3; ++i) {
    int j = tid + i * 256;
    float y = (v[i] - mean) * rinv * g[j] + bb[j];
    size_t off = t * Hdim + j;
    if (mode == 0) Out[off] = y;
    else if (sel)  Out[off] += y * sc;
  }
}

// batch-level router: mean-pool -> encoder(768->128) -> switch(128->2) -> argmax
__global__ __launch_bounds__(256) void route_kernel(
    const float* __restrict__ X, const float* __restrict__ ew, const float* __restrict__ eb,
    const float* __restrict__ sww, const float* __restrict__ swb,
    int* __restrict__ routes, float* __restrict__ pmaxB)
{
  __shared__ float pooled[Hdim];
  __shared__ float henc[Rdim];
  int bI = blockIdx.x, tid = threadIdx.x;
  for (int j = tid; j < Hdim; j += 256) {
    float s = 0.0f;
    const float* p = X + (size_t)bI * Sdim * Hdim + j;
    for (int si = 0; si < Sdim; ++si) s += p[(size_t)si * Hdim];
    pooled[j] = s * (1.0f / (float)Sdim);
  }
  __syncthreads();
  if (tid < Rdim) {
    float s = eb[tid];
    for (int j = 0; j < Hdim; ++j) s += pooled[j] * ew[j * Rdim + tid];
    henc[tid] = s;
  }
  __syncthreads();
  if (tid == 0) {
    float l0 = swb[0], l1 = swb[1];
    for (int r = 0; r < Rdim; ++r) { l0 += henc[r] * sww[r * 2 + 0]; l1 += henc[r] * sww[r * 2 + 1]; }
    float mm = fmaxf(l0, l1);
    float e0 = __expf(l0 - mm), e1 = __expf(l1 - mm);
    routes[bI] = (l1 > l0) ? 1 : 0;
    pmaxB[bI] = fmaxf(e0, e1) / (e0 + e1);
  }
}

// ---------------------------------------------------------------------------
// Host side: parse jax-pytree-flattened params (dict keys sorted) + orchestrate
// ---------------------------------------------------------------------------
namespace {
struct DenseP { const float* b; const float* w; };
struct LN2    { const float* b; const float* g; };
struct AttnP  { DenseP k; LN2 ln; const float *qa, *qb, *va, *vb; DenseP o, q, v; };
struct FfnP   { const float* b1; const float* b2; DenseP router; const float* w1; const float* w2; };
struct ExpertP { AttnP att; FfnP ffn; LN2 ln; bool lora; };
} // namespace

extern "C" void kernel_launch(void* const* d_in, const int* in_sizes, int n_in,
                              void* d_out, int out_size, void* d_ws, size_t ws_size,
                              hipStream_t stream)
{
  (void)in_sizes; (void)n_in; (void)out_size; (void)ws_size;
  const float* X    = (const float*)d_in[0];
  const float* mask = (const float*)d_in[1];

  int cur = 2;
  auto nxt = [&]() { return (const float*)d_in[cur++]; };
  auto parseDense  = [&](DenseP& d) { d.b = nxt(); d.w = nxt(); };
  auto parseLN     = [&](LN2& l)    { l.b = nxt(); l.g = nxt(); };
  auto parseAttn   = [&](AttnP& a, bool lora) {
    parseDense(a.k); parseLN(a.ln);
    if (lora) { a.qa = nxt(); a.qb = nxt(); a.va = nxt(); a.vb = nxt(); }
    else      { a.qa = a.qb = a.va = a.vb = nullptr; }
    parseDense(a.o); parseDense(a.q); parseDense(a.v);
  };
  auto parseFfn    = [&](FfnP& f) { f.b1 = nxt(); f.b2 = nxt(); parseDense(f.router); f.w1 = nxt(); f.w2 = nxt(); };
  auto parseExpert = [&](ExpertP& e, bool lora) { parseAttn(e.att, lora); parseFfn(e.ffn); parseLN(e.ln); e.lora = lora; };

  // top-level sorted keys: common, encoder, switch, unique[0], unique[1]
  ExpertP common; parseExpert(common, false);
  DenseP enc; parseDense(enc);
  DenseP sw;  parseDense(sw);
  ExpertP uq0; parseExpert(uq0, true);
  ExpertP uq1; parseExpert(uq1, true);

  // workspace carve-up
  char* wp = (char*)d_ws;
  auto balloc = [&](size_t bytes) -> void* {
    void* p = (void*)wp; wp += (bytes + 255) & ~(size_t)255; return p;
  };
  const size_t TOK = (size_t)BS;
  float* Qb   = (float*)balloc(TOK * Hdim * 4);
  float* Kb   = (float*)balloc(TOK * Hdim * 4);     // reused as O-proj output
  float* Vb   = (float*)balloc(TOK * Hdim * 4);
  float* CTXb = (float*)balloc(TOK * Hdim * 4);
  float* ATTb = (float*)balloc(TOK * Hdim * 4);
  float* FFNb = (float*)balloc(TOK * Hdim * 4);
  float* Tb   = (float*)balloc(TOK * Rdim * 4);
  float* Hcat = (float*)balloc(TOK * 4 * Rdim * 4);
  int*   routes = (int*)balloc(Bdim * 4);
  float* pmaxB  = (float*)balloc(Bdim * 4);
  int*   fidx   = (int*)balloc(TOK * 4);
  float* fpmax  = (float*)balloc(TOK * 4);

  auto gemm = [&](const float* A, int lda, const float* Bm, int ldb, float* C, int ldc,
                  const float* bias, int M, int N, int K, int flags) {
    dim3 grid(N / 128, M / 128);
    if (flags == 0 && bias)
      gemm_wmma_kernel<0, true><<<grid, 256, 0, stream>>>(A, lda, Bm, ldb, C, ldc, bias, K);
    else if (flags == 0)
      gemm_wmma_kernel<0, false><<<grid, 256, 0, stream>>>(A, lda, Bm, ldb, C, ldc, bias, K);
    else if (flags == F_ACCUM)
      gemm_wmma_kernel<F_ACCUM, false><<<grid, 256, 0, stream>>>(A, lda, Bm, ldb, C, ldc, bias, K);
    else
      gemm_wmma_kernel<F_GELU, true><<<grid, 256, 0, stream>>>(A, lda, Bm, ldb, C, ldc, bias, K);
  };

  route_kernel<<<Bdim, 256, 0, stream>>>(X, enc.w, enc.b, sw.w, sw.b, routes, pmaxB);

  auto run_expert = [&](const ExpertP& E, int mode, int eidx) {
    const int M = BS;
    gemm(X, Hdim, E.att.q.w, Hdim, Qb, Hdim, E.att.q.b, M, Hdim, Hdim, 0);
    gemm(X, Hdim, E.att.k.w, Hdim, Kb, Hdim, E.att.k.b, M, Hdim, Hdim, 0);
    gemm(X, Hdim, E.att.v.w, Hdim, Vb, Hdim, E.att.v.b, M, Hdim, Hdim, 0);
    if (E.lora) {
      gemm(X, Hdim, E.att.qa, Rdim, Tb, Rdim, nullptr, M, Rdim, Hdim, 0);
      gemm(Tb, Rdim, E.att.qb, Hdim, Qb, Hdim, nullptr, M, Hdim, Rdim, F_ACCUM);
      gemm(X, Hdim, E.att.va, Rdim, Tb, Rdim, nullptr, M, Rdim, Hdim, 0);
      gemm(Tb, Rdim, E.att.vb, Hdim, Vb, Hdim, nullptr, M, Hdim, Rdim, F_ACCUM);
    }
    attn_kernel<<<dim3(Sdim / 32, NHn, Bdim), 256, 0, stream>>>(Qb, Kb, Vb, mask, CTXb);
    gemm(CTXb, Hdim, E.att.o.w, Hdim, Kb, Hdim, E.att.o.b, M, Hdim, Hdim, 0);  // Kb = O out
    add_ln_kernel<<<BS, 256, 0, stream>>>(X, Kb, E.att.ln.g, E.att.ln.b, ATTb);
    ffn_router_kernel<<<BS / 256, 256, 0, stream>>>(ATTb, E.ffn.router.w, E.ffn.router.b, fidx, fpmax);
    for (int e = 0; e < 4; ++e)
      gemm(ATTb, Hdim, E.ffn.w1 + (size_t)e * Hdim * Rdim, Rdim,
           Hcat + e * Rdim, 4 * Rdim, E.ffn.b1 + e * Rdim, M, Rdim, Hdim, F_GELU);
    hcat_mask_kernel<<<(BS * 4 * Rdim) / 256, 256, 0, stream>>>(Hcat, fidx, fpmax);
    gemm(Hcat, 4 * Rdim, E.ffn.w2, Hdim, FFNb, Hdim, nullptr, M, Hdim, 4 * Rdim, 0);
    final_ln_combine_kernel<<<BS, 256, 0, stream>>>(ATTb, FFNb, E.ffn.b2, fidx, fpmax,
                                                    E.ln.g, E.ln.b, routes, pmaxB,
                                                    mode, eidx, (float*)d_out);
  };

  run_expert(common, /*mode=*/0, /*eidx=*/0);   // d_out = common
  run_expert(uq0,    /*mode=*/1, /*eidx=*/0);   // d_out += routed unique 0
  run_expert(uq1,    /*mode=*/1, /*eidx=*/1);   // d_out += routed unique 1
}
